// SoftDecisionTree_73555609911418
// MI455X (gfx1250) — matrix operand order repaired and checked by reading, not compiled, not run on hardware
//
#include <hip/hip_runtime.h>
#include <hip/hip_bf16.h>
#include <math.h>

typedef __attribute__((ext_vector_type(16))) _Float16 v16h;
typedef __attribute__((ext_vector_type(8)))  _Float16 v8h;
typedef __attribute__((ext_vector_type(8)))  float    v8f;
typedef __attribute__((ext_vector_type(4)))  float    v4f;
typedef __attribute__((ext_vector_type(4)))  int      v4i;

#define BATCH     8192
#define K_DIM     4096
#define N_NODES   255
#define N_PAD     256
#define N_LEAVES  256
#define N_CLASSES 10
#define TREE_DEPTH 8

#define MTILE   64
#define KTILE   32
#define ROWPADH 40   // halves per LDS row (80 bytes: 16B-aligned, bank-spreading)

// ---- CDNA5 async global->LDS path (guarded; fallback = register staging) ----
#if defined(__has_builtin)
#if __has_builtin(__builtin_amdgcn_global_load_async_to_lds_b128)
#define HAS_ASYNC_LDS 1
#endif
#endif
#ifndef HAS_ASYNC_LDS
#define HAS_ASYNC_LDS 0
#endif

#if HAS_ASYNC_LDS
// builtin signature (from clang diagnostic): (int4 AS1*, int4 AS3*, imm, imm)
typedef __attribute__((address_space(1))) v4i gmem_v4i;
typedef __attribute__((address_space(3))) v4i lds_v4i;
#endif

__device__ __forceinline__ void sdt_wait_async() {
#if HAS_ASYNC_LDS
#if __has_builtin(__builtin_amdgcn_s_wait_asynccnt)
  __builtin_amdgcn_s_wait_asynccnt(0);
#else
  asm volatile("s_wait_asynccnt 0x0" ::: "memory");
#endif
#endif
}

union Frag { v16h v; v8h h[2]; };

// ---------------------------------------------------------------------------
// Kernel 0: convert W (255x4096 f32) -> Wh (256x4096 f16, row 255 zeroed)
// ---------------------------------------------------------------------------
__global__ __launch_bounds__(256)
void sdt_wconv_kernel(const float* __restrict__ W, _Float16* __restrict__ Wh) {
  const int idx = blockIdx.x * 256 + threadIdx.x;          // 0 .. 256*4096-1
  const float v = (idx < N_NODES * K_DIM) ? W[idx] : 0.0f; // pad node 255 with 0
  Wh[idx] = (_Float16)v;
}

// ---------------------------------------------------------------------------
// Kernel 1: leaf_reward[l] = softmax(leaf_dist[l,:]) . class_reward
// ---------------------------------------------------------------------------
__global__ __launch_bounds__(256)
void sdt_leaf_kernel(const float* __restrict__ leaf_dist,
                     const float* __restrict__ class_reward,
                     float* __restrict__ lr) {
  const int t = threadIdx.x;
  if (t < N_LEAVES) {
    float v[N_CLASSES];
    float m = -1e30f;
#pragma unroll
    for (int c = 0; c < N_CLASSES; ++c) {
      v[c] = leaf_dist[t * N_CLASSES + c];
      m = fmaxf(m, v[c]);
    }
    float s = 0.0f;
#pragma unroll
    for (int c = 0; c < N_CLASSES; ++c) { v[c] = __expf(v[c] - m); s += v[c]; }
    const float inv = 1.0f / s;
    float acc = 0.0f;
#pragma unroll
    for (int c = 0; c < N_CLASSES; ++c) acc += v[c] * inv * class_reward[c];
    lr[t] = acc;
  }
}

// ---------------------------------------------------------------------------
// Kernel 2: gates[8192][256] = sigmoid(beta * (x @ W^T + b))
// WMMA f16 GEMM, M-tile 64, full N=256 per block, K-step 32.
// Double-buffered LDS, one barrier per K-step, async global->LDS for B.
// ---------------------------------------------------------------------------
__global__ __launch_bounds__(256)
void sdt_gates_gemm_kernel(const float* __restrict__ x,
                           const _Float16* __restrict__ Wh,
                           const float* __restrict__ b,
                           const float* __restrict__ beta,
                           float* __restrict__ gates) {
  __shared__ __align__(16) _Float16 lA[2][MTILE][ROWPADH];   // x tiles (f16)
  __shared__ __align__(16) _Float16 lB[2][N_PAD][ROWPADH];   // W tiles (f16)

  const int tid    = threadIdx.x;
  const int wave   = tid >> 5;
  const int lane   = tid & 31;
  const int blockM = blockIdx.x * MTILE;

  v8f acc[4][2] = {{{0.f}}};

  // cooperative-load roles: A tile is 64 rows x 32 f32; 4 threads per row
  const int arow = tid >> 2;            // 0..63
  const int acol = (tid & 3) * 8;       // 0,8,16,24

  // fragment geometry (ISA 7.12.2, wave32 f16 layouts)
  const int laneN  = lane & 15;
  const int kSel   = lane >> 4;         // 0: lanes 0-15, 1: lanes 16-31
  const int aKBase = kSel * 8;          // A: K = aKBase..+7 and aKBase+16..+23
  const int bKOff  = kSel * 8;          // B: K = bKOff..+7 and bKOff+16..+23
  const int n0     = wave * 32;         // this wave's first output column

  const float*    aSrc = x  + (size_t)(blockM + arow) * K_DIM + acol;
  const _Float16* bSrc = Wh + (size_t)tid * K_DIM;      // thread t = node row t

  // ---- prologue: stage K-tile 0 into buffer 0 ----
  {
    v4f x0 = *(const v4f*)(aSrc);
    v4f x1 = *(const v4f*)(aSrc + 4);
    _Float16* dstA = &lA[0][arow][acol];
#pragma unroll
    for (int j = 0; j < 4; ++j) dstA[j]     = (_Float16)x0[j];
#pragma unroll
    for (int j = 0; j < 4; ++j) dstA[4 + j] = (_Float16)x1[j];

#if HAS_ASYNC_LDS
    const char* g = (const char*)bSrc;
    char*       l = (char*)&lB[0][tid][0];
#pragma unroll
    for (int c = 0; c < 4; ++c)
      __builtin_amdgcn_global_load_async_to_lds_b128(
          (gmem_v4i*)(g + 16 * c), (lds_v4i*)(l + 16 * c), 0, 0);
#else
    v8h w0 = *(const v8h*)(bSrc);
    v8h w1 = *(const v8h*)(bSrc + 8);
    v8h w2 = *(const v8h*)(bSrc + 16);
    v8h w3 = *(const v8h*)(bSrc + 24);
    *(v8h*)&lB[0][tid][0]  = w0;
    *(v8h*)&lB[0][tid][8]  = w1;
    *(v8h*)&lB[0][tid][16] = w2;
    *(v8h*)&lB[0][tid][24] = w3;
#endif
  }
  sdt_wait_async();
  __syncthreads();

  int buf = 0;
  for (int kc = 0; kc < K_DIM; kc += KTILE) {
    const bool hasNext = (kc + KTILE) < K_DIM;

    // ---- issue next tile's global traffic up front (overlaps with WMMA) ----
    v4f x0 = {0.f, 0.f, 0.f, 0.f}, x1 = {0.f, 0.f, 0.f, 0.f};
#if !HAS_ASYNC_LDS
    v8h w0 = {}, w1 = {}, w2 = {}, w3 = {};
#endif
    if (hasNext) {
      const float* srcA = aSrc + kc + KTILE;
      x0 = *(const v4f*)(srcA);
      x1 = *(const v4f*)(srcA + 4);
#if HAS_ASYNC_LDS
      const char* g = (const char*)(bSrc + kc + KTILE);
      char*       l = (char*)&lB[buf ^ 1][tid][0];
#pragma unroll
      for (int c = 0; c < 4; ++c)
        __builtin_amdgcn_global_load_async_to_lds_b128(
            (gmem_v4i*)(g + 16 * c), (lds_v4i*)(l + 16 * c), 0, 0);
#else
      const _Float16* srcB = bSrc + kc + KTILE;
      w0 = *(const v8h*)(srcB);
      w1 = *(const v8h*)(srcB + 8);
      w2 = *(const v8h*)(srcB + 16);
      w3 = *(const v8h*)(srcB + 24);
#endif
      __builtin_prefetch(srcA + KTILE, 0, 1);   // global_prefetch_b8
    }

    // ---- compute on current buffer ----
    Frag afr[4], bfr[2];
#pragma unroll
    for (int nt = 0; nt < 2; ++nt) {
      const _Float16* p = &lB[buf][n0 + nt * 16 + laneN][bKOff];
      bfr[nt].h[0] = *(const v8h*)(p);
      bfr[nt].h[1] = *(const v8h*)(p + 16);
    }
#pragma unroll
    for (int mt = 0; mt < 4; ++mt) {
      const _Float16* p = &lA[buf][mt * 16 + laneN][aKBase];
      afr[mt].h[0] = *(const v8h*)(p);        // K = aKBase..aKBase+7
      afr[mt].h[1] = *(const v8h*)(p + 16);   // K = aKBase+16..aKBase+23
    }
#pragma unroll
    for (int mt = 0; mt < 4; ++mt)
#pragma unroll
      for (int nt = 0; nt < 2; ++nt)
        acc[mt][nt] = __builtin_amdgcn_wmma_f32_16x16x32_f16(
            false, afr[mt].v, false, bfr[nt].v,
            (short)0, acc[mt][nt], false, false);

    // ---- deposit staged registers into the alternate buffer ----
    if (hasNext) {
      _Float16* dstA = &lA[buf ^ 1][arow][acol];
#pragma unroll
      for (int j = 0; j < 4; ++j) dstA[j]     = (_Float16)x0[j];
#pragma unroll
      for (int j = 0; j < 4; ++j) dstA[4 + j] = (_Float16)x1[j];
#if !HAS_ASYNC_LDS
      *(v8h*)&lB[buf ^ 1][tid][0]  = w0;
      *(v8h*)&lB[buf ^ 1][tid][8]  = w1;
      *(v8h*)&lB[buf ^ 1][tid][16] = w2;
      *(v8h*)&lB[buf ^ 1][tid][24] = w3;
#endif
    }
    sdt_wait_async();
    __syncthreads();
    buf ^= 1;
  }

  // ---- epilogue: sigmoid(beta*(z+b)), C layout: VGPR r -> M=r / M=8+r ----
  const int mHi = kSel * 8;
#pragma unroll
  for (int nt = 0; nt < 2; ++nt) {
    const int col = n0 + nt * 16 + laneN;
    const float bv = (col < N_NODES) ? b[col]    : 0.0f;
    const float be = (col < N_NODES) ? beta[col] : 0.0f;
#pragma unroll
    for (int mt = 0; mt < 4; ++mt) {
#pragma unroll
      for (int r = 0; r < 8; ++r) {
        const int row = blockM + mt * 16 + mHi + r;
        const float z = be * (acc[mt][nt][r] + bv);
        gates[(size_t)row * N_PAD + col] = 1.0f / (1.0f + __expf(-z));
      }
    }
  }
}

// ---------------------------------------------------------------------------
// Kernel 3: reward[row] = sum_l (prod_d gate-or-(1-gate)) * leaf_reward[l]
// One wave per batch row; gates row staged in LDS; shuffle reduction.
// ---------------------------------------------------------------------------
__global__ __launch_bounds__(256)
void sdt_reward_kernel(const float* __restrict__ gates,
                       const float* __restrict__ lr,
                       float* __restrict__ out) {
  __shared__ float g[8][N_PAD];
  const int wave = threadIdx.x >> 5;
  const int lane = threadIdx.x & 31;
  const int row  = blockIdx.x * 8 + wave;

#pragma unroll
  for (int i = 0; i < 8; ++i)
    g[wave][lane + i * 32] = gates[(size_t)row * N_PAD + lane + i * 32];
  __syncthreads();

  float local = 0.0f;
#pragma unroll
  for (int li = 0; li < 8; ++li) {
    const int leaf = lane * 8 + li;
    float p = 1.0f;
#pragma unroll
    for (int d = 0; d < TREE_DEPTH; ++d) {
      const int node = (1 << d) - 1 + (leaf >> (TREE_DEPTH - d));
      const float gv = g[wave][node];
      p *= ((leaf >> (TREE_DEPTH - 1 - d)) & 1) ? (1.0f - gv) : gv;
    }
    local += p * lr[leaf];
  }
#pragma unroll
  for (int off = 16; off > 0; off >>= 1)
    local += __shfl_down(local, off, 32);
  if (lane == 0) out[row] = local;
}

// ---------------------------------------------------------------------------
extern "C" void kernel_launch(void* const* d_in, const int* in_sizes, int n_in,
                              void* d_out, int out_size, void* d_ws, size_t ws_size,
                              hipStream_t stream) {
  (void)in_sizes; (void)n_in; (void)out_size; (void)ws_size;
  const float* x            = (const float*)d_in[0];
  const float* W            = (const float*)d_in[1];
  const float* b            = (const float*)d_in[2];
  const float* beta         = (const float*)d_in[3];
  const float* leaf_dist    = (const float*)d_in[4];
  const float* class_reward = (const float*)d_in[5];
  float* out = (float*)d_out;

  char* ws = (char*)d_ws;
  _Float16* Wh = (_Float16*)ws;                                   // 2 MB
  float* lr    = (float*)(ws + (size_t)N_PAD * K_DIM * 2);        // 1 KB
  float* gates = (float*)(ws + (size_t)N_PAD * K_DIM * 2 + 4096); // 8 MB

  sdt_wconv_kernel<<<(N_PAD * K_DIM) / 256, 256, 0, stream>>>(W, Wh);
  sdt_leaf_kernel<<<1, 256, 0, stream>>>(leaf_dist, class_reward, lr);
  sdt_gates_gemm_kernel<<<BATCH / MTILE, 256, 0, stream>>>(x, Wh, b, beta, gates);
  sdt_reward_kernel<<<BATCH / 8, 256, 0, stream>>>(gates, lr, out);
}
